// SelectionMLP_15556371546444
// MI455X (gfx1250) — matrix operand order repaired and checked by reading, not compiled, not run on hardware
//
#include <hip/hip_runtime.h>

// ---------------- problem constants ----------------
#define Bb   512
#define INN  512
#define CBn  8
#define HDn  256
#define CHn  2048      // CB*HD
#define CCn  2560      // IN+CH
#define FC1n 1024
#define OUTn 512
#define EPSf 1e-5f

typedef _Float16 h4  __attribute__((ext_vector_type(4)));
typedef _Float16 h8  __attribute__((ext_vector_type(8)));
typedef _Float16 h16 __attribute__((ext_vector_type(16)));
typedef float    f4v __attribute__((ext_vector_type(4)));
typedef float    f8  __attribute__((ext_vector_type(8)));

// =====================================================================
// Generic WMMA GEMM:  C(MxN) = A(MxK) @ W(NxK)^T   (f32 accumulate)
// Block = 128 threads (4 waves), tile 128x64, K-step 32 via LDS.
// Each wave owns 32 rows (2 x 16-row WMMA subtiles), B-frags reused.
// Staging is two-phase (batch loads, then convert+store) so the global
// loads clause instead of serializing on s_wait_loadcnt.
// Requirements: M%128==0, K%32==0. N masked (zero-fill) if N%64!=0.
// WMODE 0: C[row*N+col] (+batch*strideC)
// WMODE 1: flat rows are (b,l) with l<Ltrans; write C[(b*N+col)*Ltrans+l]
// WMODE 2: per-batch transposed: C[batch*strideC + col*M + row]
// =====================================================================
template<int WMODE, typename AT, typename CT>
__global__ __launch_bounds__(128) void gemm_wmma(
    const AT* __restrict__ A, const float* __restrict__ W,
    CT* __restrict__ C, const float* __restrict__ bias,
    int M, int N, int K, long long strideA, long long strideC, int Ltrans)
{
  __shared__ __align__(16) _Float16 As[128 * 40];
  __shared__ __align__(16) _Float16 Bs[64 * 40];
  const int tid  = threadIdx.x;
  const int wv   = tid >> 5;
  const int lane = tid & 31;
  const long long bm = (long long)blockIdx.x * 128;
  const int       bn = blockIdx.y * 64;
  const AT* Ab = A + (long long)blockIdx.z * strideA;

  f8 acc[2][4] = {};

  const int nl = lane & 15;   // n (or m) within 16
  const int hi = lane >> 4;   // K-half selector

  for (int k0 = 0; k0 < K; k0 += 32) {
    // ---- stage A tile 128x32: load phase then store phase ----
    if (sizeof(AT) == 4) {
      const float* Gf = (const float*)Ab;
      f4v va[8];
      #pragma unroll
      for (int i = 0; i < 8; ++i) {
        int idx = tid + i * 128;
        int r = idx >> 3, c4 = (idx & 7) * 4;
        va[i] = *(const f4v*)&Gf[(bm + r) * (long long)K + (k0 + c4)];
      }
      #pragma unroll
      for (int i = 0; i < 8; ++i) {
        int idx = tid + i * 128;
        int r = idx >> 3, c4 = (idx & 7) * 4;
        h4 hv = { (_Float16)va[i][0], (_Float16)va[i][1],
                  (_Float16)va[i][2], (_Float16)va[i][3] };
        *(h4*)&As[r * 40 + c4] = hv;
      }
    } else {
      const _Float16* Gh = (const _Float16*)Ab;
      h8 va[4];
      #pragma unroll
      for (int i = 0; i < 4; ++i) {
        int idx = tid + i * 128;
        int r = idx >> 2, c8 = (idx & 3) * 8;
        va[i] = *(const h8*)&Gh[(bm + r) * (long long)K + (k0 + c8)];
      }
      #pragma unroll
      for (int i = 0; i < 4; ++i) {
        int idx = tid + i * 128;
        int r = idx >> 2, c8 = (idx & 3) * 8;
        *(h8*)&As[r * 40 + c8] = va[i];
      }
    }
    // ---- stage W tile 64x32 (rows = output channels), zero-fill n>=N ----
    {
      f4v vw[4];
      #pragma unroll
      for (int i = 0; i < 4; ++i) {
        int idx = tid + i * 128;
        int r = idx >> 3, c4 = (idx & 7) * 4;
        f4v v = {};
        if (bn + r < N) v = *(const f4v*)&W[(long long)(bn + r) * K + (k0 + c4)];
        vw[i] = v;
      }
      #pragma unroll
      for (int i = 0; i < 4; ++i) {
        int idx = tid + i * 128;
        int r = idx >> 3, c4 = (idx & 7) * 4;
        h4 hv = { (_Float16)vw[i][0], (_Float16)vw[i][1],
                  (_Float16)vw[i][2], (_Float16)vw[i][3] };
        *(h4*)&Bs[r * 40 + c4] = hv;
      }
    }
    __syncthreads();

    // A fragments (16x32 each): lane m=nl, lanes16-31 hold K 8..15 / 24..31
    h16 a[2];
    #pragma unroll
    for (int ms = 0; ms < 2; ++ms) {
      const _Float16* arow = &As[(wv * 32 + ms * 16 + nl) * 40];
      h8 alo = *(const h8*)(arow + hi * 8);
      h8 ahi = *(const h8*)(arow + 16 + hi * 8);
      a[ms] = __builtin_shufflevector(alo, ahi, 0,1,2,3,4,5,6,7,8,9,10,11,12,13,14,15);
    }

    #pragma unroll
    for (int nb = 0; nb < 4; ++nb) {
      // B fragment (32x16): lane n=nl, lanes16-31 hold K 8..15 / 24..31, B[k][n]=W[n][k]
      const _Float16* brow = &Bs[(nb * 16 + nl) * 40];
      h8 blo = *(const h8*)(brow + hi * 8);
      h8 bhi = *(const h8*)(brow + 16 + hi * 8);
      h16 b = __builtin_shufflevector(blo, bhi, 0,1,2,3,4,5,6,7,8,9,10,11,12,13,14,15);
      acc[0][nb] = __builtin_amdgcn_wmma_f32_16x16x32_f16(
          false, a[0], false, b, (short)0, acc[0][nb], false, false);
      acc[1][nb] = __builtin_amdgcn_wmma_f32_16x16x32_f16(
          false, a[1], false, b, (short)0, acc[1][nb], false, false);
    }
    __syncthreads();
  }

  // epilogue: C layout -> lane n = nl; VGPR r = row (hi*8 + r)
  #pragma unroll
  for (int ms = 0; ms < 2; ++ms) {
    #pragma unroll
    for (int nb = 0; nb < 4; ++nb) {
      int nglob = bn + nb * 16 + nl;
      if (nglob >= N) continue;
      float bv = bias ? bias[nglob] : 0.f;
      #pragma unroll
      for (int r = 0; r < 8; ++r) {
        long long grow = bm + wv * 32 + ms * 16 + hi * 8 + r;
        float val = acc[ms][nb][r] + bv;
        long long addr;
        if (WMODE == 0) {
          addr = (long long)blockIdx.z * strideC + grow * N + nglob;
        } else if (WMODE == 1) {
          long long bb = grow / Ltrans, l = grow % Ltrans;
          addr = (bb * N + nglob) * (long long)Ltrans + l;
        } else {
          addr = (long long)blockIdx.z * strideC + (long long)nglob * M + grow;
        }
        C[addr] = (CT)val;
      }
    }
  }
}

// ---------------- column stats (sum, sumsq) over rows of (R x C), C pow2 ----
template<typename T>
__global__ void colstats(const T* __restrict__ X, float* __restrict__ st,
                         long long R, int C, long long rpb)
{
  int t = threadIdx.x;
  int col, rstep;
  long long r0;
  if (C >= 256) { col = blockIdx.x * 256 + t; r0 = (long long)blockIdx.y * rpb; rstep = 1; }
  else          { col = t % C; r0 = (long long)blockIdx.y * rpb + (t / C); rstep = 256 / C; }
  long long rend = (long long)(blockIdx.y + 1) * rpb; if (rend > R) rend = R;
  float s = 0.f, s2 = 0.f;
  for (long long r = r0; r < rend; r += rstep) {
    float v = (float)X[r * C + col]; s += v; s2 += v * v;
  }
  atomicAdd(&st[col], s);
  atomicAdd(&st[C + col], s2);
}

// ---------------- BN (+optional lrelu) in place, channel-last ----------------
template<typename T>
__global__ void bn_apply(T* __restrict__ X, const float* __restrict__ st,
                         const float* __restrict__ g, const float* __restrict__ bt,
                         long long R, int C, int relu)
{
  long long n = R * (long long)C;
  long long i = (long long)blockIdx.x * 256 + threadIdx.x;
  if (i >= n) return;
  int c = (int)(i & (C - 1));                  // C is a power of two
  float invR = 1.f / (float)R;
  float m = st[c] * invR;
  float v = st[C + c] * invR - m * m;
  float y = ((float)X[i] - m) * rsqrtf(v + EPSf) * g[c] + bt[c];
  if (relu) y = y >= 0.f ? y : 0.2f * y;
  X[i] = (T)y;
}

// ---------------- stats / BN for (B, C, L) channel-mid layout ----------------
template<typename T>
__global__ void chanstats_mid(const T* __restrict__ X, float* __restrict__ st,
                              int Bn, int C, int L, int bpb)
{
  int o = blockIdx.x, t = threadIdx.x;
  float s = 0.f, s2 = 0.f;
  int bend = (blockIdx.y + 1) * bpb; if (bend > Bn) bend = Bn;
  for (int b = blockIdx.y * bpb; b < bend; ++b) {
    const T* p = X + ((long long)b * C + o) * L;
    for (int l = t; l < L; l += 256) { float v = (float)p[l]; s += v; s2 += v * v; }
  }
  __shared__ float sA[256], sB[256];
  sA[t] = s; sB[t] = s2; __syncthreads();
  for (int w = 128; w > 0; w >>= 1) {
    if (t < w) { sA[t] += sA[t + w]; sB[t] += sB[t + w]; }
    __syncthreads();
  }
  if (t == 0) { atomicAdd(&st[o], sA[0]); atomicAdd(&st[C + o], sB[0]); }
}

template<typename T>
__global__ void bn_apply_mid(T* __restrict__ X, const float* __restrict__ st,
                             const float* __restrict__ g, const float* __restrict__ bt,
                             int Bn, int C, int L, int relu)
{
  long long n = (long long)Bn * C * L;
  long long i = (long long)blockIdx.x * 256 + threadIdx.x;
  if (i >= n) return;
  int o = (int)((i / L) % C);
  float cnt = (float)Bn * (float)L;
  float m = st[o] / cnt;
  float v = st[C + o] / cnt - m * m;
  float y = ((float)X[i] - m) * rsqrtf(v + EPSf) * g[o] + bt[o];
  if (relu) y = y >= 0.f ? y : 0.2f * y;
  X[i] = (T)y;
}

// ---------------- per-row sum / sumsq of x (B x IN) ----------------
__global__ void rowstats(const float* __restrict__ x, float* __restrict__ S,
                         float* __restrict__ S2)
{
  int b = blockIdx.x, t = threadIdx.x;
  float s = 0.f, s2 = 0.f;
  for (int i = t; i < INN; i += 256) {
    float v = x[(long long)b * INN + i]; s += v; s2 += v * v;
  }
  __shared__ float sA[256], sB[256];
  sA[t] = s; sB[t] = s2; __syncthreads();
  for (int w = 128; w > 0; w >>= 1) {
    if (t < w) { sA[t] += sA[t + w]; sB[t] += sB[t + w]; }
    __syncthreads();
  }
  if (t == 0) { S[b] = sA[0]; S2[b] = sB[0]; }
}

// ---------------- global scalar sum / sumsq (for feat) ----------------
__global__ void scalarstats(const float* __restrict__ X, float* __restrict__ st, long long n)
{
  int t = threadIdx.x;
  float s = 0.f, s2 = 0.f;
  for (long long i = (long long)blockIdx.x * 256 + t; i < n; i += (long long)gridDim.x * 256) {
    float v = X[i]; s += v; s2 += v * v;
  }
  __shared__ float sA[256], sB[256];
  sA[t] = s; sB[t] = s2; __syncthreads();
  for (int w = 128; w > 0; w >>= 1) {
    if (t < w) { sA[t] += sA[t + w]; sB[t] += sB[t + w]; }
    __syncthreads();
  }
  if (t == 0) { atomicAdd(&st[0], sA[0]); atomicAdd(&st[1], sB[0]); }
}

// ---------------- gumbel softmax selection + z features ----------------
__global__ void select_feat(const float* __restrict__ logits, const float* __restrict__ gum,
                            const float* __restrict__ S, const float* __restrict__ S2,
                            float* __restrict__ feat)
{
  int idx = blockIdx.x * 256 + threadIdx.x;
  if (idx >= Bb * HDn) return;
  int b = idx >> 8, hd = idx & 255;
  float s[CBn]; float mx = -3.4e38f; int am = 0;
  #pragma unroll
  for (int cb = 0; cb < CBn; ++cb) {
    int j = b * CHn + cb * HDn + hd;
    float v = logits[j] + gum[j];
    s[cb] = v;
    if (v > mx) { mx = v; am = cb; }      // first max (jnp.argmax)
  }
  float es = 0.f;
  #pragma unroll
  for (int cb = 0; cb < CBn; ++cb) { s[cb] = expf(s[cb] - mx); es += s[cb]; }
  float inv = 1.f / es;
  float Sv = S[b], S2v = S2[b];
  #pragma unroll
  for (int cb = 0; cb < CBn; ++cb) {
    float y = s[cb] * inv;
    float hard = (cb == am) ? 1.f : 0.f;
    float w = (hard - y) + y;            // straight-through, forward order
    float xs = w * Sv, xp = w * S2v;
    feat[(long long)b * CCn + INN + cb * HDn + hd] = 0.5f * (xs * xs - xp);
  }
}

__global__ void copy_x(const float* __restrict__ x, float* __restrict__ feat)
{
  int i = blockIdx.x * 256 + threadIdx.x;
  if (i >= Bb * INN) return;
  int b = i >> 9, c = i & 511;
  feat[(long long)b * CCn + c] = x[i];
}

// e1 = lrelu(bn3(eW1 (32x1) * feat)) -- rank-1 => analytic scaled scalar stats
__global__ void e1_expand(const float* __restrict__ feat, const float* __restrict__ fst,
                          const float* __restrict__ eW1, const float* __restrict__ g,
                          const float* __restrict__ bt, _Float16* __restrict__ out)
{
  long long i = (long long)blockIdx.x * 256 + threadIdx.x;
  const long long tot = (long long)Bb * CCn * 32;
  if (i >= tot) return;
  int o = (int)(i & 31);
  long long row = i >> 5;
  const float cnt = (float)Bb * (float)CCn;
  float mf = fst[0] / cnt;
  float vf = fst[1] / cnt - mf * mf;
  float a = eW1[o];
  float y = a * (feat[row] - mf) * rsqrtf(a * a * vf + EPSf) * g[o] + bt[o];
  y = y >= 0.f ? y : 0.2f * y;
  out[i] = (_Float16)y;
}

// final d3: out[b*512+l] = dbias3 + sum_i dW3[i] * d2n[(b*512+l)*32 + i]
__global__ void d3_out(const float* __restrict__ d2n, const float* __restrict__ dW3,
                       const float* __restrict__ db3, float* __restrict__ out)
{
  int i = blockIdx.x * 256 + threadIdx.x;
  if (i >= Bb * OUTn) return;
  const float* p = d2n + (long long)i * 32;
  float s = db3[0];
  #pragma unroll
  for (int j = 0; j < 32; ++j) s += dW3[j] * p[j];
  out[i] = s;
}

// =====================================================================
extern "C" void kernel_launch(void* const* d_in, const int* in_sizes, int n_in,
                              void* d_out, int out_size, void* d_ws, size_t ws_size,
                              hipStream_t stream)
{
  (void)in_sizes; (void)n_in; (void)out_size; (void)ws_size;
  const float* x      = (const float*)d_in[0];
  const float* gumbel = (const float*)d_in[1];
  const float* cW1 = (const float*)d_in[2];
  const float* cg1 = (const float*)d_in[3];
  const float* cb1 = (const float*)d_in[4];
  const float* cW2 = (const float*)d_in[5];
  const float* cg2 = (const float*)d_in[6];
  const float* cb2 = (const float*)d_in[7];
  const float* cW3 = (const float*)d_in[8];
  const float* cg3 = (const float*)d_in[9];
  const float* cb3 = (const float*)d_in[10];
  const float* eW1 = (const float*)d_in[11];
  const float* eg1 = (const float*)d_in[12];
  const float* eb1 = (const float*)d_in[13];
  const float* eW2 = (const float*)d_in[14];
  const float* eg2 = (const float*)d_in[15];
  const float* eb2 = (const float*)d_in[16];
  const float* eW3 = (const float*)d_in[17];
  const float* eg3 = (const float*)d_in[18];
  const float* eb3 = (const float*)d_in[19];
  const float* fW1 = (const float*)d_in[20];
  const float* fg1 = (const float*)d_in[21];
  const float* fb1 = (const float*)d_in[22];
  const float* fW2 = (const float*)d_in[23];
  const float* fbias2 = (const float*)d_in[24];
  const float* dW1 = (const float*)d_in[25];
  const float* dg1 = (const float*)d_in[26];
  const float* db1 = (const float*)d_in[27];
  const float* dW2 = (const float*)d_in[28];
  const float* dg2 = (const float*)d_in[29];
  const float* db2 = (const float*)d_in[30];
  const float* dW3 = (const float*)d_in[31];
  const float* dbias3 = (const float*)d_in[32];
  float* out = (float*)d_out;

  char* ws = (char*)d_ws;
  size_t cur = 0;
  auto alloc = [&](size_t bytes) -> char* {
    char* p = ws + cur; cur += (bytes + 255) & ~(size_t)255; return p;
  };

  float* r1   = (float*)alloc((size_t)512 * 1024 * 4);
  float* r2   = (float*)alloc((size_t)512 * 1024 * 4);
  float* r3   = (float*)alloc((size_t)512 * 2048 * 4);
  float* Sb   = (float*)alloc(512 * 4);
  float* S2b  = (float*)alloc(512 * 4);
  float* fst  = (float*)alloc(2 * 4);
  float* stat = (float*)alloc(4096 * 4);
  float* feat = (float*)alloc((size_t)Bb * CCn * 4);
  _Float16* e1b = (_Float16*)alloc((size_t)Bb * CCn * 32 * 2);     //  84 MB
  _Float16* e2b = (_Float16*)alloc((size_t)Bb * CCn * 64 * 2);     // 168 MB
  _Float16* e3b = (_Float16*)alloc((size_t)Bb * 128 * CCn * 2);    // 336 MB
  _Float16* f1b = (_Float16*)alloc((size_t)Bb * 128 * 1024 * 2);   // 134 MB
  // dead-buffer aliases (lifetimes disjoint):
  float* f2T = (float*)e3b;   // 512*512*128 f32 (134MB) after e3 consumed
  float* d1b = (float*)e2b;   // 512*512*64  f32 ( 67MB) after e2 consumed
  float* d2b = (float*)e1b;   // 512*512*32  f32 ( 34MB) after e1 consumed

  const long long ME_ROWS = (long long)Bb * CCn;   // 1310720 flat (b,l) rows

  // row sums of x
  rowstats<<<dim3(Bb), dim3(256), 0, stream>>>(x, Sb, S2b);

  // ---- c-path: 3 GEMMs + BN + lrelu ----
  gemm_wmma<0, float, float><<<dim3(4, 16, 1), dim3(128), 0, stream>>>(
      x, cW1, r1, nullptr, 512, 1024, 512, 0, 0, 0);
  (void)hipMemsetAsync(stat, 0, 2 * 1024 * 4, stream);
  colstats<float><<<dim3(4, 8), dim3(256), 0, stream>>>(r1, stat, 512, 1024, 64);
  bn_apply<float><<<dim3((512 * 1024 + 255) / 256), dim3(256), 0, stream>>>(
      r1, stat, cg1, cb1, 512, 1024, 1);

  gemm_wmma<0, float, float><<<dim3(4, 16, 1), dim3(128), 0, stream>>>(
      r1, cW2, r2, nullptr, 512, 1024, 1024, 0, 0, 0);
  (void)hipMemsetAsync(stat, 0, 2 * 1024 * 4, stream);
  colstats<float><<<dim3(4, 8), dim3(256), 0, stream>>>(r2, stat, 512, 1024, 64);
  bn_apply<float><<<dim3((512 * 1024 + 255) / 256), dim3(256), 0, stream>>>(
      r2, stat, cg2, cb2, 512, 1024, 1);

  gemm_wmma<0, float, float><<<dim3(4, 32, 1), dim3(128), 0, stream>>>(
      r2, cW3, r3, nullptr, 512, 2048, 1024, 0, 0, 0);
  (void)hipMemsetAsync(stat, 0, 2 * 2048 * 4, stream);
  colstats<float><<<dim3(8, 8), dim3(256), 0, stream>>>(r3, stat, 512, 2048, 64);
  bn_apply<float><<<dim3((512 * 2048 + 255) / 256), dim3(256), 0, stream>>>(
      r3, stat, cg3, cb3, 512, 2048, 1);

  // ---- selection + feat ----
  select_feat<<<dim3((Bb * HDn) / 256), dim3(256), 0, stream>>>(r3, gumbel, Sb, S2b, feat);
  copy_x<<<dim3((Bb * INN) / 256), dim3(256), 0, stream>>>(x, feat);

  // ---- e-path ----
  (void)hipMemsetAsync(fst, 0, 8, stream);
  scalarstats<<<dim3(1024), dim3(256), 0, stream>>>(feat, fst, ME_ROWS);
  e1_expand<<<dim3((unsigned)((ME_ROWS * 32) / 256)), dim3(256), 0, stream>>>(
      feat, fst, eW1, eg1, eb1, e1b);

  gemm_wmma<0, _Float16, _Float16><<<dim3((unsigned)(ME_ROWS / 128), 1, 1), dim3(128), 0, stream>>>(
      e1b, eW2, e2b, nullptr, (int)ME_ROWS, 64, 32, 0, 0, 0);
  (void)hipMemsetAsync(stat, 0, 2 * 64 * 4, stream);
  colstats<_Float16><<<dim3(1, 320), dim3(256), 0, stream>>>(e2b, stat, ME_ROWS, 64, 4096);
  bn_apply<_Float16><<<dim3((unsigned)((ME_ROWS * 64) / 256)), dim3(256), 0, stream>>>(
      e2b, stat, eg2, eb2, ME_ROWS, 64, 1);

  gemm_wmma<1, _Float16, _Float16><<<dim3((unsigned)(ME_ROWS / 128), 2, 1), dim3(128), 0, stream>>>(
      e2b, eW3, e3b, nullptr, (int)ME_ROWS, 128, 64, 0, 0, CCn);
  (void)hipMemsetAsync(stat, 0, 2 * 128 * 4, stream);
  chanstats_mid<_Float16><<<dim3(128, 64), dim3(256), 0, stream>>>(e3b, stat, Bb, 128, CCn, 8);
  bn_apply_mid<_Float16><<<dim3((unsigned)(((long long)Bb * 128 * CCn) / 256)), dim3(256), 0, stream>>>(
      e3b, stat, eg3, eb3, Bb, 128, CCn, 1);

  // ---- f-path (batched per b) ----
  gemm_wmma<0, _Float16, _Float16><<<dim3(1, 16, Bb), dim3(128), 0, stream>>>(
      e3b, fW1, f1b, nullptr, 128, 1024, CCn, (long long)128 * CCn, (long long)128 * 1024, 0);
  (void)hipMemsetAsync(stat, 0, 2 * 1024 * 4, stream);
  colstats<_Float16><<<dim3(4, 64), dim3(256), 0, stream>>>(f1b, stat, (long long)Bb * 128, 1024, 1024);
  bn_apply<_Float16><<<dim3((unsigned)(((long long)Bb * 128 * 1024) / 256)), dim3(256), 0, stream>>>(
      f1b, stat, fg1, fb1, (long long)Bb * 128, 1024, 1);

  gemm_wmma<2, _Float16, float><<<dim3(1, 8, Bb), dim3(128), 0, stream>>>(
      f1b, fW2, f2T, fbias2, 128, 512, 1024, (long long)128 * 1024, (long long)512 * 128, 0);

  // ---- d-path (batched per b) ----
  gemm_wmma<0, float, float><<<dim3(4, 1, Bb), dim3(128), 0, stream>>>(
      f2T, dW1, d1b, nullptr, 512, 64, 128, (long long)512 * 128, (long long)512 * 64, 0);
  (void)hipMemsetAsync(stat, 0, 2 * 64 * 4, stream);
  colstats<float><<<dim3(1, 64), dim3(256), 0, stream>>>(d1b, stat, (long long)Bb * 512, 64, 4096);
  bn_apply<float><<<dim3((unsigned)(((long long)Bb * 512 * 64) / 256)), dim3(256), 0, stream>>>(
      d1b, stat, dg1, db1, (long long)Bb * 512, 64, 1);

  gemm_wmma<0, float, float><<<dim3(4, 1, Bb), dim3(128), 0, stream>>>(
      d1b, dW2, d2b, nullptr, 512, 32, 64, (long long)512 * 64, (long long)512 * 32, 0);
  (void)hipMemsetAsync(stat, 0, 2 * 32 * 4, stream);
  colstats<float><<<dim3(1, 64), dim3(256), 0, stream>>>(d2b, stat, (long long)Bb * 512, 32, 4096);
  bn_apply<float><<<dim3((unsigned)(((long long)Bb * 512 * 32) / 256)), dim3(256), 0, stream>>>(
      d2b, stat, dg2, db2, (long long)Bb * 512, 32, 1);

  d3_out<<<dim3((Bb * OUTn) / 256), dim3(256), 0, stream>>>(d2b, dW3, dbias3, out);
}
